// Self_Attn_trimap_31791347925282
// MI455X (gfx1250) — matrix operand order repaired and checked by reading, not compiled, not run on hardware
//
#include <hip/hip_runtime.h>
#include <hip/hip_bf16.h>

// ---------------------------------------------------------------------------
// Shapes: B=4, C=512, H=W=192, H4=W4=48, N=48*48=2304, C8=64, 3 branches.
// ---------------------------------------------------------------------------

typedef __attribute__((ext_vector_type(16))) __bf16 v16bf;
typedef __attribute__((ext_vector_type(8)))  float  v8f;
typedef __attribute__((ext_vector_type(4)))  unsigned int u32x4;

#define WMMA_BF16(a, b, c) \
  __builtin_amdgcn_wmma_f32_16x16x32_bf16(false, (a), false, (b), (short)0, (c), false, false)

__device__ __forceinline__ __bf16 f2bf(float x) {
  unsigned u = __builtin_bit_cast(unsigned, x);
  unsigned r = (u + 0x7FFFu + ((u >> 16) & 1u)) >> 16;
  unsigned short s = (unsigned short)r;
  return __builtin_bit_cast(__bf16, s);
}

union BFFrag { u32x4 q[2]; v16bf v; };

// A-matrix 16x32 bf16 fragment from row-major [M][K] storage (pitch in elems).
// Lane l holds row M=l%16; K values { (l/16)*8 .. +7, 16+(l/16)*8 .. +7 } (+k0).
__device__ __forceinline__ v16bf loadA(const __bf16* __restrict__ base, int row0,
                                       int pitch, int k0, int lane) {
  const __bf16* p = base + (size_t)(row0 + (lane & 15)) * pitch + k0 + ((lane >> 4) << 3);
  BFFrag f;
  f.q[0] = *(const u32x4*)(p);
  f.q[1] = *(const u32x4*)(p + 16);
  return f.v;
}

// B-matrix 32x16 bf16 fragment from [N][K] storage (K contiguous, pitch in elems).
// Lane l holds column N=l%16; K values { (l/16)*8 .. +7, 16+(l/16)*8 .. +7 } (+k0).
__device__ __forceinline__ v16bf loadB(const __bf16* __restrict__ base, int col0,
                                       int pitch, int k0, int lane) {
  const __bf16* p = base + (size_t)(col0 + (lane & 15)) * pitch + k0 + ((lane >> 4) << 3);
  BFFrag f;
  f.q[0] = *(const u32x4*)(p);
  f.q[1] = *(const u32x4*)(p + 16);
  return f.v;
}

// ---------------------------------------------------------------------------
// K0: f32 -> bf16 weight conversion
// ---------------------------------------------------------------------------
__global__ void k_cvt(const float* __restrict__ src, __bf16* __restrict__ dst, int n) {
  int i = blockIdx.x * 256 + threadIdx.x;
  if (i < n) dst[i] = f2bf(src[i]);
}

// ---------------------------------------------------------------------------
// K1: trimap 192->48 bilinear downsample (avg of the 2x2 at offsets 1,2)
// ---------------------------------------------------------------------------
__global__ void k_tri_down(const float* __restrict__ tri, float* __restrict__ t4) {
  int i = blockIdx.x * 256 + threadIdx.x;           // over 4*3*2304
  if (i >= 4 * 3 * 2304) return;
  int n = i % 2304; int bc = i / 2304;
  int h4 = n / 48, w4 = n % 48;
  const float* p = tri + ((size_t)bc * 192 + (4 * h4 + 1)) * 192 + 4 * w4 + 1;
  t4[i] = 0.25f * (p[0] + p[1] + p[192] + p[193]);
}

// ---------------------------------------------------------------------------
// K2a: x 192->48 bilinear downsample -> out4[b][c][n] (f32, also the residual)
// ---------------------------------------------------------------------------
__global__ void k_x_down(const float* __restrict__ x, float* __restrict__ out4) {
  size_t i = (size_t)blockIdx.x * 256 + threadIdx.x; // over 4*512*2304
  if (i >= (size_t)4 * 512 * 2304) return;
  int n = (int)(i % 2304); size_t bc = i / 2304;
  int h4 = n / 48, w4 = n % 48;
  const float* p = x + (bc * 192 + (4 * h4 + 1)) * 192 + 4 * w4 + 1;
  out4[i] = 0.25f * (p[0] + p[1] + p[192] + p[193]);
}

// ---------------------------------------------------------------------------
// K2b: LDS-transpose out4 (c-major) into masked bf16 activations xm[rb][n][c]
// (n-major, c contiguous — the [M][K]/[N][K] layout every GEMM below needs).
// grid (72 n-tiles, 8 c-tiles, 4 batch), 256 threads.
// ---------------------------------------------------------------------------
__global__ void __launch_bounds__(256)
k_mask_pack(const float* __restrict__ out4, const float* __restrict__ t4,
            __bf16* __restrict__ xm) {
  __shared__ float tile[64][33];
  const int n0 = blockIdx.x * 32;
  const int c0 = blockIdx.y * 64;
  const int b  = blockIdx.z;
  const int tn = threadIdx.x & 31;
  const int tc = threadIdx.x >> 5;                  // 0..7
  const float* src = out4 + ((size_t)b * 512 + c0) * 2304 + n0;
#pragma unroll
  for (int p = 0; p < 8; ++p)
    tile[tc + p * 8][tn] = src[(size_t)(tc + p * 8) * 2304 + tn];
  __syncthreads();
  const int cl = threadIdx.x & 63;
  const int nb = threadIdx.x >> 6;                  // 0..3
#pragma unroll
  for (int r = 0; r < 3; ++r) {
    const int ch = 2 - r;                           // fg->2, tr->1, bg->0
    __bf16* dst = xm + ((size_t)(r * 4 + b) * 2304 + n0) * 512 + c0;
#pragma unroll
    for (int p = 0; p < 8; ++p) {
      const int nl = nb + p * 4;
      const float m = t4[((size_t)b * 3 + ch) * 2304 + n0 + nl];
      dst[(size_t)nl * 512 + cl] = f2bf(tile[cl][nl] * m);
    }
  }
}

// ---------------------------------------------------------------------------
// K3: Q/K projection GEMMs.  Qt[rb][n][64] = xm[n][:]·Wq[o][:] + bq  (bf16)
// A = xm (shared between Q and K), B = W row-major [o][c].
// grid (144 n-tiles, 12 rb), 128 threads = 4 waves (one o-tile each).
// ---------------------------------------------------------------------------
__global__ void __launch_bounds__(128)
k_qk(const __bf16* __restrict__ xm, const __bf16* __restrict__ Wq,
     const __bf16* __restrict__ Wk, const float* __restrict__ bq,
     const float* __restrict__ bk, __bf16* __restrict__ Qt, __bf16* __restrict__ Kt) {
  const int n0   = blockIdx.x * 16;
  const int rb   = blockIdx.y;
  const int r    = rb >> 2;
  const int lane = threadIdx.x & 31;
  const int o0   = (threadIdx.x >> 5) * 16;
  const __bf16* A   = xm + (size_t)rb * 2304 * 512;
  const __bf16* WqP = Wq + (size_t)r * 64 * 512;
  const __bf16* WkP = Wk + (size_t)r * 64 * 512;
  v8f aq = {}, ak = {};
  for (int kc = 0; kc < 512; kc += 32) {
    v16bf a  = loadA(A, n0, 512, kc, lane);
    v16bf fq = loadB(WqP, o0, 512, kc, lane);
    v16bf fk = loadB(WkP, o0, 512, kc, lane);
    aq = WMMA_BF16(a, fq, aq);
    ak = WMMA_BF16(a, fk, ak);
  }
  const int   o   = o0 + (lane & 15);
  const float bqv = bq[r * 64 + o];
  const float bkv = bk[r * 64 + o];
  __bf16* Qp = Qt + (size_t)rb * 2304 * 64;
  __bf16* Kp = Kt + (size_t)rb * 2304 * 64;
#pragma unroll
  for (int g = 0; g < 8; ++g) {
    const int n = n0 + g + ((lane >> 4) << 3);
    Qp[(size_t)n * 64 + o] = f2bf(aq[g] + bqv);
    Kp[(size_t)n * 64 + o] = f2bf(ak[g] + bkv);
  }
}

// ---------------------------------------------------------------------------
// K4: V projection GEMM.  Vb[rb][c][n] = Wv[c][:]·xm[n][:] + bv  (bf16)
// grid (144 n-tiles, 8 c64-blocks, 12 rb), 128 threads = 4 waves.
// ---------------------------------------------------------------------------
__global__ void __launch_bounds__(128)
k_v(const __bf16* __restrict__ xm, const __bf16* __restrict__ Wv,
    const float* __restrict__ bv, __bf16* __restrict__ Vb) {
  const int n0   = blockIdx.x * 16;
  const int c0   = blockIdx.y * 64 + (threadIdx.x >> 5) * 16;
  const int rb   = blockIdx.z;
  const int r    = rb >> 2;
  const int lane = threadIdx.x & 31;
  const __bf16* Bm = xm + (size_t)rb * 2304 * 512;
  const __bf16* Ap = Wv + (size_t)r * 512 * 512;
  v8f acc = {};
  for (int kc = 0; kc < 512; kc += 32) {
    v16bf a  = loadA(Ap, c0, 512, kc, lane);
    v16bf bb = loadB(Bm, n0, 512, kc, lane);
    acc = WMMA_BF16(a, bb, acc);
  }
  __bf16* Vp = Vb + (size_t)rb * 512 * 2304;
  const int n = n0 + (lane & 15);
#pragma unroll
  for (int g = 0; g < 8; ++g) {
    const int c = c0 + g + ((lane >> 4) << 3);
    Vp[(size_t)c * 2304 + n] = f2bf(acc[g] + bv[r * 512 + c]);
  }
}

// ---------------------------------------------------------------------------
// K5: fused  out4[b][c][n] += sum_r gamma_r * V_r · sigmoid(Qt_r Kt_r^T)^T
// grid (72 n32-tiles, 4 batch), 256 threads = 8 waves (64 c-rows each).
// Waves 0-3 build the 32x32 sigmoid tile (2 WMMAs each) into double-buffered
// LDS; all 8 waves use it as the WMMA B-operand against streamed V tiles.
// ---------------------------------------------------------------------------
__global__ void __launch_bounds__(256)
k_attn(const __bf16* __restrict__ Qt, const __bf16* __restrict__ Kt,
       const __bf16* __restrict__ Vb, float* __restrict__ out4,
       const float* __restrict__ g_fg, const float* __restrict__ g_tr,
       const float* __restrict__ g_bg) {
  __shared__ __bf16 P[2][32][40];                   // [buf][n][m], padded pitch
  const int n0   = blockIdx.x * 32;
  const int b    = blockIdx.y;
  const int lane = threadIdx.x & 31;
  const int wav  = __builtin_amdgcn_readfirstlane((int)(threadIdx.x >> 5)); // SGPR
  const int nh   = wav & 1;
  const int mh   = (wav >> 1) & 1;

  v8f acc[4][2];
#pragma unroll
  for (int t = 0; t < 4; ++t)
#pragma unroll
    for (int h = 0; h < 2; ++h) acc[t][h] = v8f{};

  for (int r = 0; r < 3; ++r) {
    const int rb = r * 4 + b;
    const __bf16* Qp = Qt + (size_t)rb * 2304 * 64;
    const __bf16* Kp = Kt + (size_t)rb * 2304 * 64;
    const __bf16* Vp = Vb + (size_t)rb * 512 * 2304;
    const float gam = (r == 0) ? g_fg[0] : ((r == 1) ? g_tr[0] : g_bg[0]);

    // Q fragments for this workgroup's n-tile (valid rows for every wave).
    v16bf aQ0 = loadA(Qp, n0 + nh * 16, 64, 0, lane);
    v16bf aQ1 = loadA(Qp, n0 + nh * 16, 64, 32, lane);

    for (int ms = 0; ms < 72; ++ms) {
      const int m0 = ms * 32;
      if (wav < 4) {                                // scalar branch (wav is SGPR)
        v16bf bK0 = loadB(Kp, m0 + mh * 16, 64, 0, lane);
        v16bf bK1 = loadB(Kp, m0 + mh * 16, 64, 32, lane);
        v8f S = {};
        S = WMMA_BF16(aQ0, bK0, S);
        S = WMMA_BF16(aQ1, bK1, S);
        const int ml    = mh * 16 + (lane & 15);
        const int nbase = nh * 16 + ((lane >> 4) << 3);
#pragma unroll
        for (int g = 0; g < 8; ++g) {               // gamma folded into P
          float p = gam / (1.0f + __expf(-S[g]));
          P[ms & 1][nbase + g][ml] = f2bf(p);
        }
        if (ms + 1 < 72)
          __builtin_prefetch(Kp + (size_t)(m0 + 32 + mh * 16) * 64, 0, 1);
      }
      __syncthreads();
      v16bf bP0 = loadB(&P[ms & 1][0][0], 0, 40, 0, lane);
      v16bf bP1 = loadB(&P[ms & 1][0][0], 16, 40, 0, lane);
#pragma unroll
      for (int t = 0; t < 4; ++t) {
        const int c0 = wav * 64 + t * 16;
        v16bf aV = loadA(Vp, c0, 2304, m0, lane);
        acc[t][0] = WMMA_BF16(aV, bP0, acc[t][0]);
        acc[t][1] = WMMA_BF16(aV, bP1, acc[t][1]);
      }
      // double-buffered P + ds-load->WMMA waits make one barrier/step safe
    }
  }

  float* Op = out4 + (size_t)b * 512 * 2304;
#pragma unroll
  for (int t = 0; t < 4; ++t)
#pragma unroll
    for (int h = 0; h < 2; ++h) {
      const int n = n0 + h * 16 + (lane & 15);
#pragma unroll
      for (int g = 0; g < 8; ++g) {
        const int c = wav * 64 + t * 16 + g + ((lane >> 4) << 3);
        Op[(size_t)c * 2304 + n] += acc[t][h][g];   // residual already in out4
      }
    }
}

// ---------------------------------------------------------------------------
// K6: 48->192 bilinear upsample (half-pixel centers, edge clamp)
// ---------------------------------------------------------------------------
__global__ void k_upsample(const float* __restrict__ out4, float* __restrict__ out) {
  size_t i = (size_t)blockIdx.x * 256 + threadIdx.x;
  if (i >= (size_t)4 * 512 * 192 * 192) return;
  int wI = (int)(i % 192);
  int hI = (int)((i / 192) % 192);
  size_t bc = i / (192 * 192);
  float tw = 0.25f * wI - 0.375f;
  float th = 0.25f * hI - 0.375f;
  int w0 = (int)floorf(tw); float fw = tw - w0;
  int h0 = (int)floorf(th); float fh = th - h0;
  int w0c = min(max(w0, 0), 47), w1c = min(max(w0 + 1, 0), 47);
  int h0c = min(max(h0, 0), 47), h1c = min(max(h0 + 1, 0), 47);
  const float* p = out4 + bc * 2304;
  float v00 = p[h0c * 48 + w0c], v01 = p[h0c * 48 + w1c];
  float v10 = p[h1c * 48 + w0c], v11 = p[h1c * 48 + w1c];
  out[i] = (1.0f - fh) * ((1.0f - fw) * v00 + fw * v01) +
           fh * ((1.0f - fw) * v10 + fw * v11);
}

// ---------------------------------------------------------------------------
extern "C" void kernel_launch(void* const* d_in, const int* in_sizes, int n_in,
                              void* d_out, int out_size, void* d_ws, size_t ws_size,
                              hipStream_t stream) {
  (void)in_sizes; (void)n_in; (void)out_size; (void)ws_size;
  const float* x   = (const float*)d_in[0];
  const float* tri = (const float*)d_in[1];
  const float* Wq_[3] = {(const float*)d_in[2],  (const float*)d_in[9],  (const float*)d_in[16]};
  const float* bq_[3] = {(const float*)d_in[3],  (const float*)d_in[10], (const float*)d_in[17]};
  const float* Wk_[3] = {(const float*)d_in[4],  (const float*)d_in[11], (const float*)d_in[18]};
  const float* bk_[3] = {(const float*)d_in[5],  (const float*)d_in[12], (const float*)d_in[19]};
  const float* Wv_[3] = {(const float*)d_in[6],  (const float*)d_in[13], (const float*)d_in[20]};
  const float* bv_[3] = {(const float*)d_in[7],  (const float*)d_in[14], (const float*)d_in[21]};
  const float* gm_[3] = {(const float*)d_in[8],  (const float*)d_in[15], (const float*)d_in[22]};

  char* ws = (char*)d_ws;
  size_t off = 0;
  auto alloc = [&](size_t bytes) -> void* {
    off = (off + 255) & ~(size_t)255;
    void* p = ws + off;
    off += bytes;
    return p;
  };
  float*  t4   = (float*) alloc((size_t)4 * 3 * 2304 * 4);
  __bf16* xm   = (__bf16*)alloc((size_t)12 * 2304 * 512 * 2);
  __bf16* Qt   = (__bf16*)alloc((size_t)12 * 2304 * 64 * 2);
  __bf16* Kt   = (__bf16*)alloc((size_t)12 * 2304 * 64 * 2);
  __bf16* Vb   = (__bf16*)alloc((size_t)12 * 512 * 2304 * 2);
  __bf16* Wqb  = (__bf16*)alloc((size_t)3 * 64 * 512 * 2);
  __bf16* Wkb  = (__bf16*)alloc((size_t)3 * 64 * 512 * 2);
  __bf16* Wvb  = (__bf16*)alloc((size_t)3 * 512 * 512 * 2);
  float*  out4 = (float*) alloc((size_t)4 * 512 * 2304 * 4);
  float*  bqA  = (float*) alloc((size_t)3 * 64 * 4);
  float*  bkA  = (float*) alloc((size_t)3 * 64 * 4);
  float*  bvA  = (float*) alloc((size_t)3 * 512 * 4);

  for (int r = 0; r < 3; ++r) {
    k_cvt<<<(64 * 512 + 255) / 256, 256, 0, stream>>>(Wq_[r], Wqb + (size_t)r * 64 * 512, 64 * 512);
    k_cvt<<<(64 * 512 + 255) / 256, 256, 0, stream>>>(Wk_[r], Wkb + (size_t)r * 64 * 512, 64 * 512);
    k_cvt<<<(512 * 512 + 255) / 256, 256, 0, stream>>>(Wv_[r], Wvb + (size_t)r * 512 * 512, 512 * 512);
    hipMemcpyAsync(bqA + r * 64,  bq_[r], 64 * 4,  hipMemcpyDeviceToDevice, stream);
    hipMemcpyAsync(bkA + r * 64,  bk_[r], 64 * 4,  hipMemcpyDeviceToDevice, stream);
    hipMemcpyAsync(bvA + r * 512, bv_[r], 512 * 4, hipMemcpyDeviceToDevice, stream);
  }
  k_tri_down<<<(4 * 3 * 2304 + 255) / 256, 256, 0, stream>>>(tri, t4);
  k_x_down<<<(int)(((size_t)4 * 512 * 2304 + 255) / 256), 256, 0, stream>>>(x, out4);
  k_mask_pack<<<dim3(72, 8, 4), 256, 0, stream>>>(out4, t4, xm);
  k_qk<<<dim3(144, 12), 128, 0, stream>>>(xm, Wqb, Wkb, bqA, bkA, Qt, Kt);
  k_v<<<dim3(144, 8, 12), 128, 0, stream>>>(xm, Wvb, bvA, Vb);
  k_attn<<<dim3(72, 4), 256, 0, stream>>>(Qt, Kt, Vb, out4, gm_[0], gm_[1], gm_[2]);
  k_upsample<<<(int)(((size_t)4 * 512 * 192 * 192 + 255) / 256), 256, 0, stream>>>(out4, (float*)d_out);
}
